// Speller_2302102471002
// MI455X (gfx1250) — compile-verified, matrix operand order checked by reading, not run on hardware
//
#include <hip/hip_runtime.h>
#include <hip/hip_bf16.h>
#include <cstdint>
#include <cstddef>

// Problem constants (match reference)
#define BB 128
#define TT 800
#define CC 256
#define HH 512
#define EE 256
#define VV 34
#define LL 200
#define NEGV (-1e9f)

typedef __attribute__((ext_vector_type(16))) _Float16 v16h;
typedef __attribute__((ext_vector_type(8)))  _Float16 v8h;
typedef __attribute__((ext_vector_type(8)))  float    v8f;

// Load a 16x32 f16 WMMA fragment (A layout; B uses the mirrored layout with
// rows = N when the weight matrix is stored [N][K] row-major, which is exactly
// how we pack weights). Per ISA 7.12.2: lanes 0-15 hold K 0..7 & 16..23,
// lanes 16-31 hold K 8..15 & 24..31, row = lane&15. Two aligned 16B loads.
__device__ __forceinline__ v16h ld_frag(const _Float16* tile, int ld, int k0) {
  const int lane = threadIdx.x & 31;
  const int r = lane & 15, hi = lane >> 4;
  const _Float16* p = tile + (size_t)r * ld + k0 + hi * 8;
  v8h lo = *(const v8h*)(p);
  v8h hh = *(const v8h*)(p + 16);
  return __builtin_shufflevector(lo, hh, 0,1,2,3,4,5,6,7,8,9,10,11,12,13,14,15);
}

__device__ __forceinline__ float sigmoidf_(float x) { return 1.0f / (1.0f + __expf(-x)); }

// ---------------------------------------------------------------------------
// Prologue kernels (run every call; deterministic)
// ---------------------------------------------------------------------------
__global__ void f2h_kernel(const float* __restrict__ src, _Float16* __restrict__ dst, int n) {
  int i = blockIdx.x * 256 + threadIdx.x;
  if (i < n) dst[i] = (_Float16)src[i];
}

// Build fp16 weight [N=2048][K=1024] = [W_a | W_b] concat along K (each [N][512])
__global__ void catw_kernel(const float* __restrict__ Wa, const float* __restrict__ Wb,
                            _Float16* __restrict__ out, int n) {
  int i = blockIdx.x * 256 + threadIdx.x;
  if (i >= n) return;
  int k = i & 1023, row = i >> 10;
  float v = (k < HH) ? Wa[(size_t)row * HH + k] : Wb[(size_t)row * HH + (k - HH)];
  out[i] = (_Float16)v;
}

__global__ void init_state_kernel(const float* __restrict__ i0, const float* __restrict__ ic0,
                                  const float* __restrict__ i1, const float* __restrict__ ic1,
                                  float* __restrict__ h0, float* __restrict__ c0,
                                  float* __restrict__ h1, float* __restrict__ c1,
                                  _Float16* __restrict__ CATV) {
  int i = blockIdx.x * 256 + threadIdx.x;
  if (i >= BB * HH) return;
  int b = i >> 9, j = i & (HH - 1);
  h0[i] = i0[j]; c0[i] = ic0[j]; h1[i] = i1[j]; c1[i] = ic1[j];
  // fp16 copy of h1 for the step-0 query GEMM (later steps: written by LSTM1)
  CATV[(size_t)b * 768 + j] = (_Float16)i1[j];
}

// ---------------------------------------------------------------------------
// K0: query GEMM. Q[128,256] = h1_h[128,512] @ Wq_h^T + bq, once per step.
// h1_h lives in CATV[:,0:512] (packed by previous step's LSTM1 / init).
// Wq native layout [C=256][H=512] == [N][K] row-major -> direct B fragments.
// ---------------------------------------------------------------------------
__global__ __launch_bounds__(256)
void query_kernel(const _Float16* __restrict__ CATV, const _Float16* __restrict__ Wqh,
                  const float* __restrict__ bq, float* __restrict__ Q) {
  const int nt = blockIdx.x;          // 0..15 (N tiles of 16 over C=256)
  const int wave = threadIdx.x >> 5, lane = threadIdx.x & 31;
  const int m0 = wave * 16;
  v8f acc = {};
  const _Float16* At = CATV + (size_t)m0 * 768;      // reads only cols < 512
  const _Float16* Wt = Wqh + (size_t)(nt * 16) * HH;
  #pragma unroll 4
  for (int k0 = 0; k0 < HH; k0 += 32) {
    v16h a = ld_frag(At, 768, k0);
    v16h b = ld_frag(Wt, HH, k0);
    acc = __builtin_amdgcn_wmma_f32_16x16x32_f16(false, a, false, b, (short)0, acc, false, false);
  }
  const int N = lane & 15, hi = lane >> 4;
  const int nh = nt * 16 + N;
  const float bb = bq[nh];
  #pragma unroll
  for (int r = 0; r < 8; ++r)
    Q[(size_t)(m0 + r + hi * 8) * CC + nh] = acc[r] + bb;
}

// ---------------------------------------------------------------------------
// K1: attention (one block per batch element). Uses precomputed query Q[b,:],
// masked softmax energies, context; writes attn to output and packs the fp16
// A0 = [embed | context | h0] row for the LSTM0 GEMM and context into CATV.
// ---------------------------------------------------------------------------
__global__ __launch_bounds__(256)
void attn_kernel(const _Float16* __restrict__ Lh, const int* __restrict__ lengths,
                 const int* __restrict__ gt, const float* __restrict__ emb,
                 const float* __restrict__ Q,
                 const float* __restrict__ h0,
                 _Float16* __restrict__ A0, _Float16* __restrict__ CATV,
                 float* __restrict__ attn_out, int step) {
  __shared__ float sh_q[CC];
  __shared__ float sh_e[TT];
  __shared__ float sred[256];
  const int b = blockIdx.x, tid = threadIdx.x;

  if (tid < CC) sh_q[tid] = Q[(size_t)b * CC + tid];
  __syncthreads();

  const int len = lengths[b];
  float lmax = NEGV;
  for (int t = tid; t < TT; t += 256) {
    float e = NEGV;
    if (t < len) {
      const v8h* lp8 = (const v8h*)(Lh + ((size_t)b * TT + t) * CC);
      float s = 0.f;
      #pragma unroll 4
      for (int c8 = 0; c8 < CC / 8; ++c8) {
        v8h v = lp8[c8];
        #pragma unroll
        for (int j = 0; j < 8; ++j) s += (float)v[j] * sh_q[c8 * 8 + j];
      }
      e = s;
    }
    sh_e[t] = e;
    lmax = fmaxf(lmax, e);
  }
  sred[tid] = lmax; __syncthreads();
  for (int s = 128; s > 0; s >>= 1) { if (tid < s) sred[tid] = fmaxf(sred[tid], sred[tid + s]); __syncthreads(); }
  const float mx = sred[0]; __syncthreads();

  float lsum = 0.f;
  for (int t = tid; t < TT; t += 256) { float x = __expf(sh_e[t] - mx); sh_e[t] = x; lsum += x; }
  sred[tid] = lsum; __syncthreads();
  for (int s = 128; s > 0; s >>= 1) { if (tid < s) sred[tid] += sred[tid + s]; __syncthreads(); }
  const float inv = 1.0f / sred[0]; __syncthreads();

  float* ao = attn_out + ((size_t)b * LL + step) * TT;
  for (int t = tid; t < TT; t += 256) { float p = sh_e[t] * inv; sh_e[t] = p; ao[t] = p; }
  __syncthreads();

  if (tid < CC) {
    float ctx = 0.f;
    const _Float16* lp = Lh + (size_t)b * TT * CC + tid;
    for (int t = 0; t < TT; ++t) ctx += sh_e[t] * (float)lp[(size_t)t * CC];
    A0[(size_t)b * 1024 + EE + tid] = (_Float16)ctx;          // context part of x
    CATV[(size_t)b * 768 + HH + tid] = (_Float16)ctx;         // context part of cat
    const int ch = (step == 0) ? 0 : gt[(size_t)b * LL + step - 1];
    A0[(size_t)b * 1024 + tid] = (_Float16)emb[(size_t)ch * EE + tid];  // embed part
  }
  for (int j = tid; j < HH; j += 256)
    A0[(size_t)b * 1024 + (EE + CC) + j] = (_Float16)h0[(size_t)b * HH + j];
}

// ---------------------------------------------------------------------------
// K2/K3: fused LSTM step. GEMM gates = A[128,1024] @ W[2048,1024]^T(row-major
// [N][K]) via WMMA, then the cell nonlinearity + state update + fp16 repack.
// Each block owns a 16-wide h slice (all 4 gates); 8 waves cover M=128.
// ---------------------------------------------------------------------------
__global__ __launch_bounds__(256)
void lstm_step_kernel(const _Float16* __restrict__ A, const _Float16* __restrict__ W,
                      const float* __restrict__ b_ih, const float* __restrict__ b_hh,
                      float* __restrict__ cst, float* __restrict__ hst,
                      const float* __restrict__ hcopy,
                      _Float16* __restrict__ pack, int pack_ld, int pack_off_copy) {
  const int nt = blockIdx.x;                 // h-tile 0..31
  const int wave = threadIdx.x >> 5;
  const int lane = threadIdx.x & 31;
  const int m0 = wave * 16;

  v8f acc0 = {}, acc1 = {}, acc2 = {}, acc3 = {};
  const _Float16* At  = A + (size_t)m0 * 1024;
  const _Float16* W0t = W + (size_t)(0 * HH + nt * 16) * 1024;  // i
  const _Float16* W1t = W + (size_t)(1 * HH + nt * 16) * 1024;  // f
  const _Float16* W2t = W + (size_t)(2 * HH + nt * 16) * 1024;  // g
  const _Float16* W3t = W + (size_t)(3 * HH + nt * 16) * 1024;  // o

  #pragma unroll 4
  for (int k0 = 0; k0 < 1024; k0 += 32) {
    v16h a  = ld_frag(At, 1024, k0);
    v16h b0 = ld_frag(W0t, 1024, k0);
    v16h b1 = ld_frag(W1t, 1024, k0);
    v16h b2 = ld_frag(W2t, 1024, k0);
    v16h b3 = ld_frag(W3t, 1024, k0);
    acc0 = __builtin_amdgcn_wmma_f32_16x16x32_f16(false, a, false, b0, (short)0, acc0, false, false);
    acc1 = __builtin_amdgcn_wmma_f32_16x16x32_f16(false, a, false, b1, (short)0, acc1, false, false);
    acc2 = __builtin_amdgcn_wmma_f32_16x16x32_f16(false, a, false, b2, (short)0, acc2, false, false);
    acc3 = __builtin_amdgcn_wmma_f32_16x16x32_f16(false, a, false, b3, (short)0, acc3, false, false);
  }

  const int N = lane & 15, hi = lane >> 4;
  const int nh = nt * 16 + N;
  const float bi = b_ih[nh]          + b_hh[nh];
  const float bf = b_ih[HH + nh]     + b_hh[HH + nh];
  const float bg = b_ih[2 * HH + nh] + b_hh[2 * HH + nh];
  const float bo = b_ih[3 * HH + nh] + b_hh[3 * HH + nh];

  #pragma unroll
  for (int r = 0; r < 8; ++r) {
    const int brow = m0 + r + hi * 8;
    const size_t si = (size_t)brow * HH + nh;
    float iv = sigmoidf_(acc0[r] + bi);
    float fv = sigmoidf_(acc1[r] + bf);
    float gv = tanhf(acc2[r] + bg);
    float ov = sigmoidf_(acc3[r] + bo);
    float cn = fv * cst[si] + iv * gv;
    float hn = ov * tanhf(cn);
    cst[si] = cn;
    hst[si] = hn;
    pack[(size_t)brow * pack_ld + nh] = (_Float16)hn;
    if (pack_off_copy >= 0)
      pack[(size_t)brow * pack_ld + pack_off_copy + nh] = (_Float16)hcopy[si];
  }
}

// ---------------------------------------------------------------------------
// K4: hid = leaky_relu(CATV[128,768] @ W1[512,768]^T + b1), WMMA.
// ---------------------------------------------------------------------------
__global__ __launch_bounds__(256)
void mlp_kernel(const _Float16* __restrict__ A, const _Float16* __restrict__ W,
                const float* __restrict__ b1, float* __restrict__ hid) {
  const int nt = blockIdx.x;           // 0..31
  const int wave = threadIdx.x >> 5, lane = threadIdx.x & 31;
  const int m0 = wave * 16;
  v8f acc = {};
  const _Float16* At = A + (size_t)m0 * 768;
  const _Float16* Wt = W + (size_t)(nt * 16) * 768;
  #pragma unroll 4
  for (int k0 = 0; k0 < 768; k0 += 32) {
    v16h a = ld_frag(At, 768, k0);
    v16h b = ld_frag(Wt, 768, k0);
    acc = __builtin_amdgcn_wmma_f32_16x16x32_f16(false, a, false, b, (short)0, acc, false, false);
  }
  const int N = lane & 15, hi = lane >> 4;
  const int nh = nt * 16 + N;
  const float bb = b1[nh];
  #pragma unroll
  for (int r = 0; r < 8; ++r) {
    const int brow = m0 + r + hi * 8;
    float v = acc[r] + bb;
    hid[(size_t)brow * HH + nh] = (v > 0.f) ? v : 0.2f * v;
  }
}

// ---------------------------------------------------------------------------
// K5: logits = hid @ W_out^T + b_out (tiny: V=34), one block per batch row.
// ---------------------------------------------------------------------------
__global__ __launch_bounds__(64)
void out_kernel(const float* __restrict__ hid, const float* __restrict__ Wo,
                const float* __restrict__ bo, float* __restrict__ logits_out, int step) {
  __shared__ float sh[HH];
  const int b = blockIdx.x, tid = threadIdx.x;
  for (int j = tid; j < HH; j += 64) sh[j] = hid[(size_t)b * HH + j];
  __syncthreads();
  if (tid < VV) {
    const float* wr = Wo + (size_t)tid * HH;
    float s = bo[tid];
    #pragma unroll 8
    for (int k = 0; k < HH; ++k) s += wr[k] * sh[k];
    logits_out[((size_t)b * LL + step) * VV + tid] = s;
  }
}

// ---------------------------------------------------------------------------
extern "C" void kernel_launch(void* const* d_in, const int* in_sizes, int n_in,
                              void* d_out, int out_size, void* d_ws, size_t ws_size,
                              hipStream_t stream) {
  const float* listener = (const float*)d_in[0];
  const int*   lengths  = (const int*)d_in[1];
  const int*   gt       = (const int*)d_in[2];
  const float* emb      = (const float*)d_in[3];
  const float* W_ih0    = (const float*)d_in[4];
  const float* W_hh0    = (const float*)d_in[5];
  const float* b_ih0    = (const float*)d_in[6];
  const float* b_hh0    = (const float*)d_in[7];
  const float* W_ih1    = (const float*)d_in[8];
  const float* W_hh1    = (const float*)d_in[9];
  const float* b_ih1    = (const float*)d_in[10];
  const float* b_hh1    = (const float*)d_in[11];
  const float* inith0   = (const float*)d_in[12];
  const float* initc0   = (const float*)d_in[13];
  const float* inith1   = (const float*)d_in[14];
  const float* initc1   = (const float*)d_in[15];
  const float* Wq       = (const float*)d_in[16];
  const float* bq       = (const float*)d_in[17];
  const float* W1       = (const float*)d_in[18];
  const float* b1       = (const float*)d_in[19];
  const float* W_out    = (const float*)d_in[20];
  const float* b_out    = (const float*)d_in[21];

  // Workspace layout (~64 MB total), 256B aligned slices.
  uint8_t* wsb = (uint8_t*)d_ws;
  size_t off = 0;
  auto alloc = [&](size_t bytes) -> void* {
    void* p = wsb + off;
    off += (bytes + 255) & ~(size_t)255;
    return p;
  };
  _Float16* Lh   = (_Float16*)alloc(sizeof(_Float16) * (size_t)BB * TT * CC); // 52.4 MB
  _Float16* W0c  = (_Float16*)alloc(sizeof(_Float16) * 2048 * 1024);          // 4 MB
  _Float16* W1c  = (_Float16*)alloc(sizeof(_Float16) * 2048 * 1024);          // 4 MB
  _Float16* W1o  = (_Float16*)alloc(sizeof(_Float16) * (size_t)HH * 768);     // 0.75 MB
  _Float16* Wqh  = (_Float16*)alloc(sizeof(_Float16) * (size_t)CC * HH);      // 0.25 MB
  _Float16* A0   = (_Float16*)alloc(sizeof(_Float16) * (size_t)BB * 1024);
  _Float16* A1   = (_Float16*)alloc(sizeof(_Float16) * (size_t)BB * 1024);
  _Float16* CATV = (_Float16*)alloc(sizeof(_Float16) * (size_t)BB * 768);
  float* h0  = (float*)alloc(sizeof(float) * (size_t)BB * HH);
  float* c0  = (float*)alloc(sizeof(float) * (size_t)BB * HH);
  float* h1  = (float*)alloc(sizeof(float) * (size_t)BB * HH);
  float* c1  = (float*)alloc(sizeof(float) * (size_t)BB * HH);
  float* hid = (float*)alloc(sizeof(float) * (size_t)BB * HH);
  float* Qb  = (float*)alloc(sizeof(float) * (size_t)BB * CC);

  float* out_logits = (float*)d_out;                                  // [B,L,V]
  float* out_attn   = out_logits + (size_t)BB * LL * VV;              // [B,L,T]

  // Prologue: fp16 conversions + packed weights + state init (every call).
  int n;
  n = BB * TT * CC;  f2h_kernel<<<(n + 255) / 256, 256, 0, stream>>>(listener, Lh, n);
  n = 2048 * 1024;   catw_kernel<<<(n + 255) / 256, 256, 0, stream>>>(W_ih0, W_hh0, W0c, n);
  n = 2048 * 1024;   catw_kernel<<<(n + 255) / 256, 256, 0, stream>>>(W_ih1, W_hh1, W1c, n);
  n = HH * 768;      f2h_kernel<<<(n + 255) / 256, 256, 0, stream>>>(W1, W1o, n);
  n = CC * HH;       f2h_kernel<<<(n + 255) / 256, 256, 0, stream>>>(Wq, Wqh, n);
  n = BB * HH;       init_state_kernel<<<(n + 255) / 256, 256, 0, stream>>>(
                         inith0, initc0, inith1, initc1, h0, c0, h1, c1, CATV);

  // 200 strictly sequential decoder steps (graph-captured kernel chain).
  for (int step = 0; step < LL; ++step) {
    // Q = h1_h @ Wq^T + bq  (h1_h is CATV[:,0:512] from prev step / init)
    query_kernel<<<16, 256, 0, stream>>>(CATV, Wqh, bq, Qb);
    attn_kernel<<<BB, 256, 0, stream>>>(Lh, lengths, gt, emb, Qb,
                                        h0, A0, CATV, out_attn, step);
    // LSTM0: A0=[embed|ctx|h0]; packs h0n -> A1[:,0:512], h1 -> A1[:,512:1024]
    lstm_step_kernel<<<32, 256, 0, stream>>>(A0, W0c, b_ih0, b_hh0, c0, h0,
                                             h1, A1, 1024, 512);
    // LSTM1: A1=[h0n|h1]; packs h1n -> CATV[:,0:512] (ctx already at 512:768)
    lstm_step_kernel<<<32, 256, 0, stream>>>(A1, W1c, b_ih1, b_hh1, c1, h1,
                                             nullptr, CATV, 768, -1);
    mlp_kernel<<<32, 256, 0, stream>>>(CATV, W1o, b1, hid);
    out_kernel<<<BB, 64, 0, stream>>>(hid, W_out, b_out, out_logits, step);
  }
}
